// MLACausalSelfAttention_80126909874599
// MI455X (gfx1250) — matrix operand order repaired and checked by reading, not compiled, not run on hardware
//
#include <hip/hip_runtime.h>
#include <hip/hip_bf16.h>

// ---------------- problem constants (match reference) ----------------
#define Bc    2
#define Tc    2048
#define Cc    2048
#define Hc    16
#define HDc   128
#define DCc   672
#define DRc   64
#define DNc   64          // HD - DR
#define GATEc 32
#define Mrows (Bc * Tc)   // 4096

typedef __bf16 bf16;
typedef __attribute__((ext_vector_type(16))) __bf16 v16bf;
typedef __attribute__((ext_vector_type(8)))  __bf16 v8bf;
typedef __attribute__((ext_vector_type(4)))  __bf16 v4bf;
typedef __attribute__((ext_vector_type(8)))  float  v8f;
typedef __attribute__((ext_vector_type(4)))  int    v4i;

static __device__ __forceinline__ bf16 f2bf(float f) {
  unsigned u = __builtin_bit_cast(unsigned, f);
  unsigned r = (u + 0x7FFFu + ((u >> 16) & 1u)) >> 16;
  unsigned short s = (unsigned short)r;
  return __builtin_bit_cast(bf16, s);
}
static __device__ __forceinline__ float bf2f(bf16 v) {
  unsigned u = ((unsigned)__builtin_bit_cast(unsigned short, v)) << 16;
  return __builtin_bit_cast(float, u);
}
static __device__ __forceinline__ v16bf cat8(v8bf lo, v8bf hi) {
  return __builtin_shufflevector(lo, hi, 0,1,2,3,4,5,6,7,8,9,10,11,12,13,14,15);
}

// ---------------- CDNA5 async global->LDS copy (ASYNCcnt) ----------------
#if defined(__gfx1250__) && \
    __has_builtin(__builtin_amdgcn_global_load_async_to_lds_b128) && \
    __has_builtin(__builtin_amdgcn_s_wait_asynccnt)
#define HAS_ASYNC 1
#else
#define HAS_ASYNC 0
#endif

#if HAS_ASYNC
static __device__ __forceinline__ void async_ld16(void* lds, const void* gp) {
  __builtin_amdgcn_global_load_async_to_lds_b128(
      (__attribute__((address_space(1))) v4i*)gp,
      (__attribute__((address_space(3))) v4i*)lds,
      0, 0);
}
#define WAIT_ASYNC(n) __builtin_amdgcn_s_wait_asynccnt(n)
#else
static __device__ __forceinline__ void async_ld16(void* lds, const void* gp) {
  *(v8bf*)lds = *(const v8bf*)gp;   // sync fallback
}
#define WAIT_ASYNC(n)
#endif

// =====================================================================
// f32 -> bf16 copy-convert (vectorized x4). n must be a multiple of 4.
// =====================================================================
__global__ __launch_bounds__(256) void convert_bf16(
    const float* __restrict__ s, bf16* __restrict__ d, int n4)
{
  int i = blockIdx.x * 256 + threadIdx.x;
  if (i < n4) {
    float4 v = ((const float4*)s)[i];
    v4bf o;
    o[0] = f2bf(v.x); o[1] = f2bf(v.y); o[2] = f2bf(v.z); o[3] = f2bf(v.w);
    *(v4bf*)(d + (size_t)i * 4) = o;
  }
}

// =====================================================================
// f32 [R][C] -> bf16 [C][R] tiled transpose-convert (32x32 tiles).
// =====================================================================
__global__ __launch_bounds__(256) void transpose_bf16(
    const float* __restrict__ src, bf16* __restrict__ dst, int R, int C)
{
  __shared__ float tile[32][33];
  const int c0 = blockIdx.x * 32, r0 = blockIdx.y * 32;
  const int tx = threadIdx.x & 31, ty = threadIdx.x >> 5;
#pragma unroll
  for (int i = ty; i < 32; i += 8) {
    int r = r0 + i, c = c0 + tx;
    tile[i][tx] = (r < R && c < C) ? src[(size_t)r * C + c] : 0.0f;
  }
  __syncthreads();
#pragma unroll
  for (int i = ty; i < 32; i += 8) {
    int c = c0 + i, r = r0 + tx;
    if (c < C && r < R) dst[(size_t)c * R + r] = f2bf(tile[tx][i]);
  }
}

// =====================================================================
// GEMM: C[M,N] = A[M,K] @ B[K,N]; A bf16 [M][K], Bt = B^T bf16 [N][K].
// Block tile 128x64, K-step 32, double-buffered async LDS staging.
// 256 threads = 8 waves; waves 4 in M x 2 in N -> 2x2 WMMA tiles each.
// Requires: M % 128 == 0, K % 32 == 0 (true for all call sites).
// =====================================================================
#define BM 128
#define BN 64
#define BK 32

template <bool OUT_BF16>
__global__ __launch_bounds__(256) void gemm_bf16_wmma(
    const bf16* __restrict__ A, const bf16* __restrict__ Bt,
    void* __restrict__ Cout, int M, int N, int K)
{
  __shared__ alignas(16) bf16 As[2][BM][BK + 8];   // 80B pitch
  __shared__ alignas(16) bf16 Bs[2][BN][BK + 8];   // column-major tile

  const int tid  = threadIdx.x;
  const int lane = tid & 31;
  const int wid  = tid >> 5;
  const int half = lane >> 4;
  const int l15  = lane & 15;

  const int wm = (wid >> 1) * 32;
  const int wn = (wid & 1) * 32;

  const int block_row = blockIdx.y * BM;
  const int block_col = blockIdx.x * BN;

  // per-thread staging coordinates (3 async b128 per thread per stage)
  const int ar0 = tid >> 2, aseg = tid & 3;             // A rep0: rows 0..63
  const int ar1 = (tid + 256) >> 2;                     // A rep1: rows 64..127
  const int bn  = block_col + (tid >> 2);
  const int bnc = (bn < N) ? bn : (N - 1);              // clamp (OOB cols unused)

  v8f acc[2][2];
#pragma unroll
  for (int mi = 0; mi < 2; ++mi)
#pragma unroll
    for (int ni = 0; ni < 2; ++ni)
#pragma unroll
      for (int j = 0; j < 8; ++j) acc[mi][ni][j] = 0.0f;

  auto stage = [&](int buf, int kk) {
    async_ld16(&As[buf][ar0][aseg * 8],
               A + (size_t)(block_row + ar0) * K + kk + aseg * 8);
    async_ld16(&As[buf][ar1][aseg * 8],
               A + (size_t)(block_row + ar1) * K + kk + aseg * 8);
    async_ld16(&Bs[buf][tid >> 2][aseg * 8],
               Bt + (size_t)bnc * K + kk + aseg * 8);
  };

  stage(0, 0);
  int cur = 0;
  for (int k0 = 0; k0 < K; k0 += BK) {
    if (k0 + BK < K) {
      stage(cur ^ 1, k0 + BK);   // overlap next tile's DMA with compute
      WAIT_ASYNC(3);             // in-order: current tile resident
    } else {
      WAIT_ASYNC(0);
    }
    __syncthreads();

    v16bf bfrag[2];
#pragma unroll
    for (int ni = 0; ni < 2; ++ni) {
      const bf16* p = &Bs[cur][wn + ni * 16 + l15][half * 16];
      bfrag[ni] = cat8(*(const v8bf*)p, *(const v8bf*)(p + 8));
    }
#pragma unroll
    for (int mi = 0; mi < 2; ++mi) {
      const bf16* p = &As[cur][wm + mi * 16 + l15][0];
      v16bf afrag = cat8(*(const v8bf*)(p + half * 16),
                         *(const v8bf*)(p + half * 16 + 8));
#pragma unroll
      for (int ni = 0; ni < 2; ++ni) {
        acc[mi][ni] = __builtin_amdgcn_wmma_f32_16x16x32_bf16(
            false, afrag, false, bfrag[ni], (short)0, acc[mi][ni], false, false);
      }
    }
    __syncthreads();
    cur ^= 1;
  }

  // ---- epilogue ----
#pragma unroll
  for (int mi = 0; mi < 2; ++mi)
#pragma unroll
    for (int ni = 0; ni < 2; ++ni)
#pragma unroll
      for (int j = 0; j < 8; ++j) {
        int r = block_row + wm + mi * 16 + j + 8 * half;
        int c = block_col + wn + ni * 16 + l15;
        if (c < N) {
          if (OUT_BF16) ((bf16*)Cout)[(size_t)r * N + c] = f2bf(acc[mi][ni][j]);
          else          ((float*)Cout)[(size_t)r * N + c] = acc[mi][ni][j];
        }
      }
}

// =====================================================================
// Prep: gate, RoPE(k_rope,q_rope), v = v + gate*ve, RMS-norm q/k.
// Q is pre-scaled by 1/sqrt(HD) (folded out of the attention loop).
// Emits bf16 Q,K at [(b*T+t)*H + h]*HD + d and V transposed at
// [((b*H + h)*HD + d)*T + t]. One block (128 threads) per (h,t,b).
// =====================================================================
__global__ __launch_bounds__(128) void mla_prep(
    const float* __restrict__ x, const float* __restrict__ ve,
    const float* __restrict__ cosb, const float* __restrict__ sinb,
    const float* __restrict__ wg,
    const bf16* __restrict__ kvfull, const bf16* __restrict__ krb,
    const bf16* __restrict__ qfull,
    bf16* __restrict__ Qn, bf16* __restrict__ Kn, bf16* __restrict__ Vt)
{
  __shared__ float red[HDc];
  __shared__ float gsh;

  const int h = blockIdx.x, t = blockIdx.y, bb = blockIdx.z;
  const int d = threadIdx.x;
  const int row = bb * Tc + t;

  if (d < GATEc) red[d] = x[(size_t)row * Cc + d] * wg[d * Hc + h];
  __syncthreads();
  if (d == 0) {
    float a = 0.0f;
#pragma unroll
    for (int i = 0; i < GATEc; ++i) a += red[i];
    gsh = 2.0f / (1.0f + __expf(-a));
  }
  __syncthreads();
  const float gate = gsh;

  const float* cs = cosb + (size_t)row * (DRc / 2);
  const float* sn = sinb + (size_t)row * (DRc / 2);

  float kd;
  if (d < DNc) {
    kd = bf2f(kvfull[(size_t)row * (Hc * (DNc + HDc)) + h * (DNc + HDc) + d]);
  } else {
    const bf16* kr = krb + (size_t)row * DRc;
    int i = d - DNc;
    if (i < DRc / 2)
      kd = bf2f(kr[i]) * cs[i] + bf2f(kr[i + DRc / 2]) * sn[i];
    else {
      int j = i - DRc / 2;
      kd = -bf2f(kr[j]) * sn[j] + bf2f(kr[j + DRc / 2]) * cs[j];
    }
  }
  const bf16* qr = qfull + (size_t)row * (Hc * HDc) + h * HDc;
  float qd;
  if (d < DNc) qd = bf2f(qr[d]);
  else {
    int i = d - DNc;
    if (i < DRc / 2)
      qd = bf2f(qr[DNc + i]) * cs[i] + bf2f(qr[DNc + i + DRc / 2]) * sn[i];
    else {
      int j = i - DRc / 2;
      qd = -bf2f(qr[DNc + j]) * sn[j] + bf2f(qr[DNc + j + DRc / 2]) * cs[j];
    }
  }

  const float eps = 1.1920929e-07f;
  __syncthreads();
  red[d] = kd * kd; __syncthreads();
  for (int s = 64; s > 0; s >>= 1) { if (d < s) red[d] += red[d + s]; __syncthreads(); }
  const float kn = kd * __frsqrt_rn(red[0] / (float)HDc + eps);
  __syncthreads();
  red[d] = qd * qd; __syncthreads();
  for (int s = 64; s > 0; s >>= 1) { if (d < s) red[d] += red[d + s]; __syncthreads(); }
  const float qscale = 0.08838834764831845f;  // 1/sqrt(128), folded into Q
  const float qn = qd * __frsqrt_rn(red[0] / (float)HDc + eps) * qscale;

  const float vv =
      bf2f(kvfull[(size_t)row * (Hc * (DNc + HDc)) + h * (DNc + HDc) + DNc + d]) +
      gate * ve[(size_t)row * (Hc * HDc) + h * HDc + d];

  const size_t o = ((size_t)row * Hc + h) * HDc + d;
  Kn[o] = f2bf(kn);
  Qn[o] = f2bf(qn);
  Vt[((size_t)(bb * Hc + h) * HDc + d) * Tc + t] = f2bf(vv);
}

// =====================================================================
// Flash attention (causal). Block = 128 q rows, 8 waves x 16 rows.
// Key blocks of 32, double-buffered async K/V staging.
// =====================================================================
__global__ __launch_bounds__(256) void mla_attn(
    const bf16* __restrict__ Q, const bf16* __restrict__ Kn,
    const bf16* __restrict__ Vt, bf16* __restrict__ Y)
{
  __shared__ alignas(16) bf16 Ks[2][32][HDc + 8];    // [key][d]
  __shared__ alignas(16) bf16 Vs[2][HDc][32 + 8];    // [d][key]
  __shared__ alignas(16) bf16 Ps[8][16][32 + 8];     // per-wave P tile

  const int tid  = threadIdx.x;
  const int lane = tid & 31;
  const int wid  = tid >> 5;
  const int half = lane >> 4;
  const int l15  = lane & 15;

  const int b  = blockIdx.z, h = blockIdx.y;
  const int q0 = blockIdx.x * 128;
  const int qbase = q0 + wid * 16;

  // Q fragments (scale pre-folded in prep): 4 d-chunks straight from global
  v16bf qf[4];
  {
    const bf16* qrow = Q + ((size_t)(b * Tc + qbase + l15) * Hc + h) * HDc;
#pragma unroll
    for (int c = 0; c < 4; ++c)
      qf[c] = cat8(*(const v8bf*)(qrow + c * 32 + half * 16),
                   *(const v8bf*)(qrow + c * 32 + half * 16 + 8));
  }

  float m[8], l[8];
  v8f o[8];
#pragma unroll
  for (int j = 0; j < 8; ++j) { m[j] = -1e30f; l[j] = 0.0f; }
#pragma unroll
  for (int c = 0; c < 8; ++c)
#pragma unroll
    for (int j = 0; j < 8; ++j) o[c][j] = 0.0f;

  const bf16* VtBH = Vt + (size_t)(b * Hc + h) * HDc * Tc;
  const bf16* KnB  = Kn + (size_t)b * Tc * Hc * HDc + (size_t)h * HDc;

  // per-thread staging coordinates (4 async b128 per thread per stage)
  const int kr0 = tid >> 4,  ksg = tid & 15;   // K: rows 0..15 / 16..31
  const int vr0 = tid >> 2,  vsg = tid & 3;    // V: rows 0..63 / 64..127

  auto stage_kv = [&](int buf, int kstart) {
    async_ld16(&Ks[buf][kr0][ksg * 8],
               KnB + (size_t)(kstart + kr0) * (Hc * HDc) + ksg * 8);
    async_ld16(&Ks[buf][kr0 + 16][ksg * 8],
               KnB + (size_t)(kstart + kr0 + 16) * (Hc * HDc) + ksg * 8);
    async_ld16(&Vs[buf][vr0][vsg * 8],
               VtBH + (size_t)vr0 * Tc + kstart + vsg * 8);
    async_ld16(&Vs[buf][vr0 + 64][vsg * 8],
               VtBH + (size_t)(vr0 + 64) * Tc + kstart + vsg * 8);
  };

  const int nkb = (q0 + 128) / 32;
  stage_kv(0, 0);
  int cur = 0;
  for (int kb = 0; kb < nkb; ++kb) {
    const int kstart = kb * 32;
    if (kb + 1 < nkb) {
      stage_kv(cur ^ 1, kstart + 32);
      WAIT_ASYNC(4);
    } else {
      WAIT_ASYNC(0);
    }
    __syncthreads();

    if (kstart <= qbase + 15) {
      v8f s[2];
#pragma unroll
      for (int ni = 0; ni < 2; ++ni)
#pragma unroll
        for (int j = 0; j < 8; ++j) s[ni][j] = 0.0f;

      // scores = Q @ K^T
#pragma unroll
      for (int c = 0; c < 4; ++c)
#pragma unroll
        for (int ni = 0; ni < 2; ++ni) {
          const bf16* p = &Ks[cur][ni * 16 + l15][c * 32 + half * 16];
          v16bf bk = cat8(*(const v8bf*)p, *(const v8bf*)(p + 8));
          s[ni] = __builtin_amdgcn_wmma_f32_16x16x32_bf16(
              false, qf[c], false, bk, (short)0, s[ni], false, false);
        }

      // causal mask
#pragma unroll
      for (int ni = 0; ni < 2; ++ni)
#pragma unroll
        for (int j = 0; j < 8; ++j) {
          int qi = qbase + j + 8 * half;
          int ki = kstart + ni * 16 + l15;
          s[ni][j] = (ki <= qi) ? s[ni][j] : -1e30f;
        }

      // online softmax (reductions within 16-lane half-groups)
      float mnew[8];
#pragma unroll
      for (int j = 0; j < 8; ++j) {
        float v = fmaxf(s[0][j], s[1][j]);
#pragma unroll
        for (int msk = 1; msk < 16; msk <<= 1)
          v = fmaxf(v, __shfl_xor(v, msk, 32));
        mnew[j] = fmaxf(m[j], v);
      }
#pragma unroll
      for (int ni = 0; ni < 2; ++ni)
#pragma unroll
        for (int j = 0; j < 8; ++j) s[ni][j] = __expf(s[ni][j] - mnew[j]);
#pragma unroll
      for (int j = 0; j < 8; ++j) {
        float v = s[0][j] + s[1][j];
#pragma unroll
        for (int msk = 1; msk < 16; msk <<= 1) v += __shfl_xor(v, msk, 32);
        float alpha = __expf(m[j] - mnew[j]);
        l[j] = l[j] * alpha + v;
        m[j] = mnew[j];
#pragma unroll
        for (int c = 0; c < 8; ++c) o[c][j] *= alpha;
      }

      // P: C-layout -> A-layout via per-wave LDS
#pragma unroll
      for (int ni = 0; ni < 2; ++ni)
#pragma unroll
        for (int j = 0; j < 8; ++j)
          Ps[wid][j + 8 * half][ni * 16 + l15] = f2bf(s[ni][j]);

      const bf16* pp = &Ps[wid][l15][0];
      v16bf pf = cat8(*(const v8bf*)(pp + half * 16),
                      *(const v8bf*)(pp + half * 16 + 8));

      // O += P @ V
#pragma unroll
      for (int c = 0; c < 8; ++c) {
        const bf16* p = &Vs[cur][c * 16 + l15][half * 16];
        v16bf bv = cat8(*(const v8bf*)p, *(const v8bf*)(p + 8));
        o[c] = __builtin_amdgcn_wmma_f32_16x16x32_bf16(
            false, pf, false, bv, (short)0, o[c], false, false);
      }
    }
    __syncthreads();
    cur ^= 1;
  }

  // epilogue: y[b,t,h,:] (bf16, feeds final GEMM)
#pragma unroll
  for (int j = 0; j < 8; ++j) {
    const float inv = __frcp_rn(l[j]);
    const int qi = qbase + j + 8 * half;
#pragma unroll
    for (int c = 0; c < 8; ++c)
      Y[(size_t)(b * Tc + qi) * (Hc * HDc) + h * HDc + c * 16 + l15] =
          f2bf(o[c][j] * inv);
  }
}

// =====================================================================
extern "C" void kernel_launch(void* const* d_in, const int* in_sizes, int n_in,
                              void* d_out, int out_size, void* d_ws, size_t ws_size,
                              hipStream_t stream) {
  const float* x       = (const float*)d_in[0];
  const float* ve      = (const float*)d_in[1];
  const float* cosb    = (const float*)d_in[2];
  const float* sinb    = (const float*)d_in[3];
  const float* w_dkv   = (const float*)d_in[4];
  const float* w_ukv   = (const float*)d_in[5];
  const float* w_kr    = (const float*)d_in[6];
  const float* w_dq    = (const float*)d_in[7];
  const float* w_q     = (const float*)d_in[8];
  const float* w_cproj = (const float*)d_in[9];
  const float* w_gate  = (const float*)d_in[10];
  float* out = (float*)d_out;

  char* ws = (char*)d_ws;
  size_t off = 0;
  auto alloc = [&](size_t bytes) -> void* {
    void* p = ws + off;
    off = (off + bytes + 255) & ~(size_t)255;
    return p;
  };

  const int KV = Hc * (DNc + HDc);  // 3072
  bf16* xb   = (bf16*)alloc((size_t)Mrows * Cc * 2);
  bf16* wdkT = (bf16*)alloc((size_t)DCc * Cc * 2);
  bf16* wukT = (bf16*)alloc((size_t)KV * DCc * 2);
  bf16* wkrT = (bf16*)alloc((size_t)DRc * Cc * 2);
  bf16* wdqT = (bf16*)alloc((size_t)DCc * Cc * 2);
  bf16* wqT  = (bf16*)alloc((size_t)(Hc * HDc) * DCc * 2);
  bf16* wcpT = (bf16*)alloc((size_t)Cc * (Hc * HDc) * 2);
  bf16* ckv  = (bf16*)alloc((size_t)Mrows * DCc * 2);      // c_kv, reused for c_q
  bf16* kvf  = (bf16*)alloc((size_t)Mrows * KV * 2);       // kv, reused for y
  bf16* krb  = (bf16*)alloc((size_t)Mrows * DRc * 2);
  bf16* qful = (bf16*)alloc((size_t)Mrows * Hc * HDc * 2);
  bf16* Qn   = (bf16*)alloc((size_t)Mrows * Hc * HDc * 2);
  bf16* Kn   = (bf16*)alloc((size_t)Mrows * Hc * HDc * 2);
  bf16* Vt   = (bf16*)alloc((size_t)Mrows * Hc * HDc * 2);
  bf16* yb   = kvf;

  {
    int n4 = Mrows * Cc / 4;
    convert_bf16<<<(n4 + 255) / 256, 256, 0, stream>>>(x, xb, n4);
  }
  auto tr = [&](const float* s, bf16* d, int R, int C) {
    transpose_bf16<<<dim3((C + 31) / 32, (R + 31) / 32), 256, 0, stream>>>(s, d, R, C);
  };
  tr(w_dkv,   wdkT, Cc, DCc);
  tr(w_ukv,   wukT, DCc, KV);
  tr(w_kr,    wkrT, Cc, DRc);
  tr(w_dq,    wdqT, Cc, DCc);
  tr(w_q,     wqT,  DCc, Hc * HDc);
  tr(w_cproj, wcpT, Hc * HDc, Cc);

  dim3 blk(256);
  auto gemmB = [&](const bf16* A, const bf16* Bt, bf16* Cm, int M, int N, int K) {
    dim3 g((N + BN - 1) / BN, M / BM);
    gemm_bf16_wmma<true><<<g, blk, 0, stream>>>(A, Bt, Cm, M, N, K);
  };

  gemmB(xb,  wdkT, ckv,  Mrows, DCc,       Cc);   // c_kv
  gemmB(ckv, wukT, kvf,  Mrows, KV,        DCc);  // kv
  gemmB(xb,  wkrT, krb,  Mrows, DRc,       Cc);   // k_rope (pre-rope)
  gemmB(xb,  wdqT, ckv,  Mrows, DCc,       Cc);   // c_q (reuse ckv)
  gemmB(ckv, wqT,  qful, Mrows, Hc * HDc,  DCc);  // q full

  mla_prep<<<dim3(Hc, Tc, Bc), 128, 0, stream>>>(
      x, ve, cosb, sinb, w_gate, kvf, krb, qful, Qn, Kn, Vt);

  mla_attn<<<dim3(Tc / 128, Hc, Bc), 256, 0, stream>>>(Qn, Kn, Vt, yb);

  {
    dim3 g((Cc + BN - 1) / BN, Mrows / BM);
    gemm_bf16_wmma<false><<<g, blk, 0, stream>>>(yb, wcpT, out, Mrows, Cc, Hc * HDc);
  }
}
